// LowRankConv3D_49039936586099
// MI455X (gfx1250) — matrix operand order repaired and checked by reading, not compiled, not run on hardware
//
#include <hip/hip_runtime.h>

// ---------------------------------------------------------------------------
// CP-decomposed 3D conv, fused single kernel for gfx1250 (MI455X).
//   Stage 1: channel->rank GEMM via v_wmma_f32_16x16x32_bf16
//   Stage 2: 27-tap separable conv (precomputed rank-wise tap products), VALU
//   Stage 3: rank->C_out GEMM via v_wmma_f32_16x16x32_bf16 (+bias)
// Intermediates live in LDS; t1 stores rank-PAIRS interleaved per dword so
// the conv stage reads ds_load_b32 and GEMM1 writes packed cvt_pk_bf16 dwords.
// ---------------------------------------------------------------------------

typedef __bf16        v16bf __attribute__((ext_vector_type(16)));
typedef __bf16        v2bf  __attribute__((ext_vector_type(2)));
typedef float         v8f   __attribute__((ext_vector_type(8)));
typedef unsigned int  v8u   __attribute__((ext_vector_type(8)));

#define WMMA_BF16(A, B, C) \
  __builtin_amdgcn_wmma_f32_16x16x32_bf16(false, (A), false, (B), (short)0, (C), false, false)

constexpr int C_IN    = 32;
constexpr int C_OUT   = 64;
constexpr int RNK     = 64;
constexpr int SPATIAL = 64 * 64 * 64;   // per (b, channel)
constexpr int TW      = 4;              // w-tile per workgroup
constexpr int DL      = 66;             // padded d-line length (dword stride in t1)

// LDS layout (bytes). t1: 18 lines x 32 rank-pairs x 66 dwords.
constexpr int OFF_TAPS = 0;                                  // 27*64*4 = 6912
constexpr int OFF_T1   = 8192;                               // 18*32*66*4 = 152064
constexpr int OFF_T3   = OFF_T1;                             // 64*256*4 = 65536 (alias, t1 dead)
constexpr int OFF_T2   = OFF_T1 + 18 * 32 * DL * 4;          // 160256; 256*64*2 = 32768
constexpr int LDS_BYTES = OFF_T2 + 256 * 64 * 2;             // 193024

__global__ __launch_bounds__(256)
void LowRankConv3D_cpd_fused(const float* __restrict__ x,
                             const float* __restrict__ ukh,
                             const float* __restrict__ ukw,
                             const float* __restrict__ ukd,
                             const float* __restrict__ ucin,
                             const float* __restrict__ ucout,
                             const float* __restrict__ bias,
                             float* __restrict__ out)
{
  extern __shared__ __align__(16) char smem[];
  float*        taps = (float*)(smem + OFF_TAPS);
  unsigned int* t1u  = (unsigned int*)(smem + OFF_T1);   // [18][32][66] rank-pair dwords
  unsigned int* t2u  = (unsigned int*)(smem + OFF_T2);   // [256][32] rank-pair dwords
  __bf16*       t2h  = (__bf16*)(smem + OFF_T2);         // same memory, bf16 view
  float*        t3   = (float*)(smem + OFF_T3);          // [64 co][256 pos]

  const int tid  = threadIdx.x;
  const int lane = tid & 31;          // wave32
  const int wave = tid >> 5;          // 8 waves
  const int row  = lane & 15;         // A-row / D-column index
  const int hi   = lane >> 4;         // half-wave select (K-split)

  const int bid = blockIdx.x;         // = b*1024 + h0*16 + wtile
  const int w0  = (bid & 15) * TW;
  const int h0  = (bid >> 4) & 63;
  const int b   = bid >> 10;

  // ---- phase 0a: per-rank tap products into LDS --------------------------
  for (int i = tid; i < 27 * 64; i += 256) {
    const int r = i & 63, t = i >> 6;
    const int kd = t % 3, kw = (t / 3) % 3, kh = t / 9;
    taps[i] = ukh[kh * 64 + r] * ukw[kw * 64 + r] * ukd[kd * 64 + r];
  }

  // ---- phase 0b: A-frags of U_c_in^T (16 ranks x 32 channels, bf16) ------
  // 16-bit A layout: lanes<16 hold K=0..7,16..23; lanes>=16 hold K=8..15,24..31
  v16bf a1[4];
  #pragma unroll
  for (int rt = 0; rt < 4; ++rt) {
    const int r = rt * 16 + row;
    #pragma unroll
    for (int j = 0; j < 8; ++j) {
      const int k = ((j < 4) ? 2 * j : 8 + 2 * j) + hi * 8;
      a1[rt][2 * j]     = (__bf16)ucin[k * RNK + r];
      a1[rt][2 * j + 1] = (__bf16)ucin[(k + 1) * RNK + r];
    }
  }

  // ---- phase 0c: B-frags of U_c_out (32 ranks x 16 couts per chunk) ------
  // 16-bit B layout: lanes<16 hold K=0..7,16..23; lanes>=16 hold K=8..15,24..31 (2/VGPR)
  v16bf b2[2][4];
  #pragma unroll
  for (int kc = 0; kc < 2; ++kc)
    #pragma unroll
    for (int ct = 0; ct < 4; ++ct) {
      const int co = ct * 16 + row;
      #pragma unroll
      for (int j = 0; j < 8; ++j) {
        const int r = kc * 32 + ((j < 4) ? 2 * j : 8 + 2 * j) + hi * 8;
        b2[kc][ct][2 * j]     = (__bf16)ucout[r * C_OUT + co];
        b2[kc][ct][2 * j + 1] = (__bf16)ucout[(r + 1) * C_OUT + co];
      }
    }

  __syncthreads();

  // ---- phase 1: GEMM1 (channels -> ranks) over padded tile ---------------
  // 18 (h,w)-lines x 5 d-groups (overlapping bases keep every lane valid).
  const int gbase[5] = {0, 16, 32, 48, 50};
  const float* xb0 = x + (size_t)b * C_IN * SPATIAL;
  for (int task = wave; task < 90; task += 8) {
    const int line = task / 5, g = task % 5;
    const int ph = line / 6, pw = line % 6;
    const int hg = h0 - 1 + ph;
    const int wg = w0 - 1 + pw;
    const int pd = gbase[g] + row;         // padded d index, 0..65
    const int dg = pd - 1;
    const bool inb = ((unsigned)hg < 64u) & ((unsigned)wg < 64u) & ((unsigned)dg < 64u);
    const int hc = inb ? hg : 0, wc = inb ? wg : 0, dc = inb ? dg : 0;
    const float* xp = xb0 + hc * 4096 + wc * 64 + dc;

    // B-frag: x, K=channels, N=positions.  Pack bf16 pairs, select dwords.
    v8u bu;
    #pragma unroll
    for (int j = 0; j < 8; ++j) {
      const int c = ((j < 4) ? 2 * j : 8 + 2 * j) + hi * 8;
      v2bf p;
      p.x = (__bf16)xp[(size_t)c * SPATIAL];          // coalesced across lanes (d)
      p.y = (__bf16)xp[(size_t)(c + 1) * SPATIAL];
      const unsigned int u = __builtin_bit_cast(unsigned int, p);
      bu[j] = inb ? u : 0u;                           // one cndmask_b32 per dword
    }
    const v16bf bx = __builtin_bit_cast(v16bf, bu);

    const v8f z = {0.f, 0.f, 0.f, 0.f, 0.f, 0.f, 0.f, 0.f};
    v8f acc[4];
    acc[0] = WMMA_BF16(a1[0], bx, z);
    acc[1] = WMMA_BF16(a1[1], bx, z);
    acc[2] = WMMA_BF16(a1[2], bx, z);
    acc[3] = WMMA_BF16(a1[3], bx, z);

    // D layout: VGPR v -> rank rt*16 + hi*8 + v; pack rank pairs into dwords.
    unsigned int* t1l = t1u + (size_t)line * 32 * DL + pd;
    #pragma unroll
    for (int rt = 0; rt < 4; ++rt)
      #pragma unroll
      for (int u = 0; u < 4; ++u) {
        const int rp = rt * 8 + hi * 4 + u;
        v2bf p;
        p.x = (__bf16)acc[rt][2 * u];
        p.y = (__bf16)acc[rt][2 * u + 1];
        t1l[rp * DL] = __builtin_bit_cast(unsigned int, p);
      }
  }
  __syncthreads();

  // ---- phase 2: 27-tap conv, rank pairs per dword (ds_load_b32) ----------
  {
    const int wl = tid >> 6;   // 0..3 (local w)
    const int d  = tid & 63;
    #pragma unroll 2
    for (int rp = 0; rp < 32; ++rp) {
      float s0 = 0.f, s1 = 0.f;
      #pragma unroll
      for (int kh = 0; kh < 3; ++kh)
        #pragma unroll
        for (int kw = 0; kw < 3; ++kw) {
          const unsigned int* tp = t1u + (size_t)((kh * 6 + wl + kw) * 32 + rp) * DL + d;
          const float*        wp = taps + ((kh * 3 + kw) * 3) * 64 + 2 * rp;
          #pragma unroll
          for (int kd = 0; kd < 3; ++kd) {
            const v2bf pr = __builtin_bit_cast(v2bf, tp[kd]);
            s0 += wp[kd * 64]     * (float)pr.x;
            s1 += wp[kd * 64 + 1] * (float)pr.y;
          }
        }
      v2bf o;
      o.x = (__bf16)s0;
      o.y = (__bf16)s1;
      t2u[tid * 32 + rp] = __builtin_bit_cast(unsigned int, o);
    }
  }
  __syncthreads();

  // ---- phase 3: GEMM2 (ranks -> C_out) + bias ----------------------------
  float bv[4];
  #pragma unroll
  for (int ct = 0; ct < 4; ++ct) bv[ct] = bias[ct * 16 + row];

  struct P32 { uint4 a, b; };
  for (int m = wave; m < 16; m += 8) {     // 16 position-tiles of 16
    const int pos = m * 16 + row;
    const __bf16* tp = t2h + pos * 64;
    // A-frag from LDS: lanes<16 need r 0..7 & 16..23; lanes>=16 r 8..15 & 24..31
    const uint4 q0 = *(const uint4*)(tp + hi * 8);
    const uint4 q1 = *(const uint4*)(tp + 16 + hi * 8);
    const uint4 q2 = *(const uint4*)(tp + 32 + hi * 8);
    const uint4 q3 = *(const uint4*)(tp + 48 + hi * 8);
    const P32 pk0 = {q0, q1};
    const P32 pk1 = {q2, q3};
    const v16bf aK0 = __builtin_bit_cast(v16bf, pk0);
    const v16bf aK1 = __builtin_bit_cast(v16bf, pk1);

    #pragma unroll
    for (int ct = 0; ct < 4; ++ct) {
      v8f c = {bv[ct], bv[ct], bv[ct], bv[ct], bv[ct], bv[ct], bv[ct], bv[ct]};
      c = WMMA_BF16(aK0, b2[0][ct], c);
      c = WMMA_BF16(aK1, b2[1][ct], c);
      // D: lane column co = ct*16+row, VGPR v -> pos m*16 + hi*8 + v
      float* t3p = t3 + (ct * 16 + row) * 256 + m * 16 + hi * 8;
      const float4 lo = {c[0], c[1], c[2], c[3]};
      const float4 hv = {c[4], c[5], c[6], c[7]};
      *(float4*)(t3p)     = lo;
      *(float4*)(t3p + 4) = hv;
    }
  }
  __syncthreads();

  // ---- phase 4: coalesced global store (t3 is [co][pos], pos contiguous) -
  float* ob = out + (size_t)b * C_OUT * SPATIAL + h0 * 4096 + w0 * 64;
  for (int i = tid; i < 64 * 64; i += 256) {
    const int co = i >> 6;
    const int p4 = (i & 63) * 4;
    const float4 v = *(const float4*)(t3 + co * 256 + p4);
    *(float4*)(ob + (size_t)co * SPATIAL + p4) = v;
  }
}

extern "C" void kernel_launch(void* const* d_in, const int* in_sizes, int n_in,
                              void* d_out, int out_size, void* d_ws, size_t ws_size,
                              hipStream_t stream) {
  const float* x     = (const float*)d_in[0];
  const float* ukh   = (const float*)d_in[1];
  const float* ukw   = (const float*)d_in[2];
  const float* ukd   = (const float*)d_in[3];
  const float* ucin  = (const float*)d_in[4];
  const float* ucout = (const float*)d_in[5];
  const float* bias  = (const float*)d_in[6];
  float* outp        = (float*)d_out;

  // grid: B(2) * H(64) * W/TW(16) = 2048 workgroups, 256 threads (8 wave32)
  LowRankConv3D_cpd_fused<<<dim3(2048), dim3(256), LDS_BYTES, stream>>>(
      x, ukh, ukw, ukd, ucin, ucout, bias, outp);
}
